// Model_31980326486320
// MI455X (gfx1250) — matrix operand order repaired and checked by reading, not compiled, not run on hardware
//
#include <hip/hip_runtime.h>
#include <hip/hip_bf16.h>

typedef __attribute__((ext_vector_type(16))) _Float16 v16h;
typedef __attribute__((ext_vector_type(8)))  float    v8f;

constexpr int Bc  = 64;
constexpr int Nc  = 2048;
constexpr int Mc  = 12;
constexpr int AIN = 64;
constexpr int AT  = 25;
constexpr int BD  = 50;

// ---------------------------------------------------------------------------
// helpers: raw-hardware activation math (v_exp_f32 / v_log_f32 / v_rcp_f32,
// no denorm-guard expansions: log argument is always in [1,2])
// ---------------------------------------------------------------------------
#define LOG2E 1.4426950408889634f
#define LN2   0.6931471805599453f

__device__ __forceinline__ float softplusf(float x) {
  // stable: max(x,0) + ln(1 + 2^(-|x|*log2e))
  float t = __builtin_amdgcn_exp2f(-fabsf(x) * LOG2E);
  return fmaf(__builtin_amdgcn_logf(1.f + t), LN2, fmaxf(x, 0.f));
}
__device__ __forceinline__ float sigmoidf(float x) {
  return __builtin_amdgcn_rcpf(1.f + __builtin_amdgcn_exp2f(-x * LOG2E));
}

// wave-local LDS producer->consumer ordering (LDS is in-order per wave;
// this pins the compiler + drains DScnt)
__device__ __forceinline__ void wave_sync_lds() {
  __builtin_amdgcn_wave_barrier();
  asm volatile("s_wait_dscnt 0x0" ::: "memory");
  __builtin_amdgcn_wave_barrier();
}

__device__ __forceinline__ v8f wmma_f16(v16h a, v16h b, v8f c) {
  return __builtin_amdgcn_wmma_f32_16x16x32_f16(false, a, false, b,
                                                (short)0, c, false, false);
}

// Store one A-matrix element (row m 0..15, feature column c) into the LDS
// fragment-swizzled layout so each lane later reads its v16h contiguously.
// 16-bit A 16x32 layout (ISA 7.12.2): lane = m + 16*((c%32>>3)&1),
// elem e = ((c%32 & 16)>>1) | (c%32 & 7), k-step = c/32.
__device__ __forceinline__ void st_a_sw(_Float16* dst, int m, int c, _Float16 v) {
  int ks = c >> 5;
  int kp = c & 31;
  int g  = (kp >> 3) & 1;
  int e  = ((kp & 16) >> 1) | (kp & 7);
  int ln = (m & 15) + (g << 4);
  dst[(ks * 32 + ln) * 16 + e] = v;
}

// Stage a [kr x nc] f32 weight (row-major, K-major like the reference's
// x @ W) into fragment-swizzled f16 LDS, zero-padded to KS*32 x NT*16.
// B 32x16 layout: lanes 0-15 hold K=0..7,16..23; lanes 16-31 K=8..15,24..31.
__device__ __forceinline__ void stage_weight_sw(_Float16* dst, const float* W,
                                                int kr, int nc, int KS, int NT,
                                                int tid, int nthr) {
  int total = KS * NT * 32 * 16;
  for (int idx = tid; idx < total; idx += nthr) {
    int e  = idx & 15;
    int l  = (idx >> 4) & 31;
    int f  = idx >> 9;  // frag id = ks*NT + nt
    int ks = f / NT;
    int nt = f - ks * NT;
    int k  = ks * 32 + ((e & 8) ? 16 : 0) + ((l & 16) ? 8 : 0) + (e & 7);
    int n  = nt * 16 + (l & 15);
    float v = (k < kr && n < nc) ? W[k * nc + n] : 0.f;
    dst[idx] = (_Float16)v;
  }
}

// ---------------------------------------------------------------------------
// Kernel 1: grouped atom-type linear  at = softplus(atom @ W_{ta|tb} + b)
// one wave = one 16-row tile; K=64 (2 ksteps), N=25 -> 2 ntiles
// ---------------------------------------------------------------------------
__global__ void __launch_bounds__(256)
k_atomtype(const float* __restrict__ atom,
           const float* __restrict__ Wta, const float* __restrict__ bta,
           const float* __restrict__ Wtb, const float* __restrict__ btb,
           float* __restrict__ at_out) {
  __shared__ __align__(32) _Float16 sW[2][2 * 2 * 32 * 16];  // 2 weights, 4KB each
  __shared__ __align__(32) _Float16 sA[8][2 * 32 * 16];      // 2KB per wave

  int tid  = threadIdx.x;
  int lane = tid & 31;
  int wv   = tid >> 5;

  stage_weight_sw(sW[0], Wta, AIN, AT, 2, 2, tid, 256);
  stage_weight_sw(sW[1], Wtb, AIN, AT, 2, 2, tid, 256);
  __syncthreads();

  int row0 = (blockIdx.x * 8 + wv) * 16;     // flat row in [0, B*N)
  int n0   = row0 % Nc;                      // 1024 % 16 == 0: no straddle
  int grp  = (n0 >= Nc / 2) ? 1 : 0;

  _Float16* aSt = sA[wv];
  for (int idx = lane; idx < 16 * AIN; idx += 32) {
    int m = idx >> 6, c = idx & 63;
    st_a_sw(aSt, m, c, (_Float16)atom[(size_t)(row0 + m) * AIN + c]);
  }
  wave_sync_lds();

  v8f acc[2] = {};
#pragma unroll
  for (int ks = 0; ks < 2; ++ks) {
    v16h a = *(const v16h*)(aSt + (ks * 32 + lane) * 16);
#pragma unroll
    for (int nt = 0; nt < 2; ++nt) {
      v16h b = *(const v16h*)(sW[grp] + ((ks * 2 + nt) * 32 + lane) * 16);
      acc[nt] = wmma_f16(a, b, acc[nt]);
    }
  }

  const float* bias = grp ? btb : bta;
  int nn = lane & 15, hb = lane >> 4;
#pragma unroll
  for (int nt = 0; nt < 2; ++nt) {
    int col = nt * 16 + nn;
    if (col < AT) {
      float bv = bias[col];
#pragma unroll
      for (int v = 0; v < 8; ++v) {
        int m = v + hb * 8;
        at_out[(size_t)(row0 + m) * AT + col] = softplusf(acc[nt][v] + bv);
      }
    }
  }
}

// ---------------------------------------------------------------------------
// Kernel 2: AtomConv. One wave = 4 atoms (= 48 edges = 3 exact WMMA tiles).
// z = [self|nbr|bond] @ W_ac + b_ac ; atom_update = sp(at + sum_m sig(g)*sp(c))
// K = 100 -> 128 (4 ksteps), N = 50 -> 4 ntiles
// ---------------------------------------------------------------------------
__global__ void __launch_bounds__(256)
k_atomconv(const float* __restrict__ at, const float* __restrict__ bond,
           const int* __restrict__ adj, const float* __restrict__ Wac,
           const float* __restrict__ bac, float* __restrict__ au_out) {
  __shared__ __align__(32) _Float16     sW[4 * 4 * 32 * 16];  // 16KB
  __shared__ __align__(32) unsigned char sBuf[8][4096];       // stage / z overlay
  __shared__ int sNbr[8][48];

  int tid  = threadIdx.x;
  int lane = tid & 31;
  int wv   = tid >> 5;

  stage_weight_sw(sW, Wac, 2 * AT + BD, 2 * AT, 4, 4, tid, 256);
  __syncthreads();

  int unit = blockIdx.x * 8 + wv;  // 4 atoms per unit
  int g0   = unit * 4;
  int b    = g0 / Nc;
  int n0   = g0 % Nc;
  size_t bN = (size_t)b * Nc;

  for (int r = lane; r < 48; r += 32) {
    int la = r / 12, m = r - la * 12;
    sNbr[wv][r] = adj[(bN + n0 + la) * Mc + m];
  }
  wave_sync_lds();

  _Float16* stage = (_Float16*)sBuf[wv];  // 16x128 f16 A tile (swizzled) = 4KB
  float*    zbuf  = (float*)sBuf[wv];     // overlay: 16x50 z, pitch 52 = 3.3KB

  float sred[4] = {0.f, 0.f, 0.f, 0.f};
  int nn = lane & 15, hb = lane >> 4;

  for (int t = 0; t < 3; ++t) {
    int r0 = t * 16;
    // stage 16 edges x 128 features (gather self/nbr/bond, pad with 0)
    for (int idx = lane; idx < 16 * 128; idx += 32) {
      int m = idx >> 7, c = idx & 127;
      int r = r0 + m;
      int la = r / 12, mm = r - la * 12;
      float v;
      if (c < 25)        v = at[(bN + n0 + la) * AT + c];
      else if (c < 50)   v = at[(bN + sNbr[wv][r]) * AT + (c - 25)];
      else if (c < 100)  v = bond[((bN + n0 + la) * Mc + mm) * BD + (c - 50)];
      else               v = 0.f;
      st_a_sw(stage, m, c, (_Float16)v);
    }
    wave_sync_lds();

    v8f acc[4] = {};
#pragma unroll
    for (int ks = 0; ks < 4; ++ks) {
      v16h a = *(const v16h*)(stage + (ks * 32 + lane) * 16);
#pragma unroll
      for (int nt = 0; nt < 4; ++nt) {
        v16h bb = *(const v16h*)(sW + ((ks * 4 + nt) * 32 + lane) * 16);
        acc[nt] = wmma_f16(a, bb, acc[nt]);
      }
    }
    wave_sync_lds();  // A reads drained before overwriting with z

    // z (+bias) -> zbuf
#pragma unroll
    for (int nt = 0; nt < 4; ++nt) {
      int col = nt * 16 + nn;
      if (col < BD) {
        float bv = bac[col];
#pragma unroll
        for (int v = 0; v < 8; ++v)
          zbuf[(v + hb * 8) * 52 + col] = acc[nt][v] + bv;
      }
    }
    wave_sync_lds();

    // partial gated reduction over this sub-tile's rows
#pragma unroll
    for (int i = 0; i < 4; ++i) {
      int oidx = lane + i * 32;  // (atom-in-unit, out-col) pairs: 4*25 = 100
      if (oidx < 100) {
        int la = oidx / 25, c = oidx - la * 25;
        int lo = la * 12, hi2 = lo + 12;
        lo  = lo  < r0      ? r0      : lo;
        hi2 = hi2 > r0 + 16 ? r0 + 16 : hi2;
        float s = 0.f;
        for (int rr = lo; rr < hi2; ++rr) {
          float gte = zbuf[(rr - r0) * 52 + c];
          float cor = zbuf[(rr - r0) * 52 + c + 25];
          s += sigmoidf(gte) * softplusf(cor);
        }
        sred[i] += s;
      }
    }
    wave_sync_lds();  // done with zbuf before next staging pass
  }

#pragma unroll
  for (int i = 0; i < 4; ++i) {
    int oidx = lane + i * 32;
    if (oidx < 100) {
      int la = oidx / 25, c = oidx - la * 25;
      float av = at[(bN + n0 + la) * AT + c];
      au_out[(bN + n0 + la) * AT + c] = softplusf(av + sred[i]);
    }
  }
}

// ---------------------------------------------------------------------------
// Kernel 3: fused Embedding + BondConv. One wave = 16 edges.
// d = sp([a_i|a_j] @ W_emb + b)  (K=50->64);  out = sp([d|bond] @ W_bc + b)
// ---------------------------------------------------------------------------
__global__ void __launch_bounds__(128)
k_bond(const float* __restrict__ au, const float* __restrict__ bond,
       const int* __restrict__ tup, const float* __restrict__ Wemb,
       const float* __restrict__ bemb, const float* __restrict__ Wbc,
       const float* __restrict__ bbc, float* __restrict__ out_bond) {
  __shared__ __align__(32) _Float16 sWe[2 * 4 * 32 * 16];  // 8KB
  __shared__ __align__(32) _Float16 sWb[4 * 4 * 32 * 16];  // 16KB
  __shared__ __align__(32) _Float16 sA1[4][2 * 32 * 16];   // 2KB / wave
  __shared__ __align__(32) _Float16 sA2[4][4 * 32 * 16];   // 4KB / wave
  __shared__ int sT[4][32];

  int tid  = threadIdx.x;
  int lane = tid & 31;
  int wv   = tid >> 5;

  stage_weight_sw(sWe, Wemb, 2 * AT, BD, 2, 4, tid, 128);
  stage_weight_sw(sWb, Wbc, 2 * BD, BD, 4, 4, tid, 128);
  __syncthreads();

  int tile = blockIdx.x * 4 + wv;
  size_t e0 = (size_t)tile * 16;                 // N*M = 24576 divisible by 16
  int   bI  = (int)(e0 / ((size_t)Nc * Mc));
  size_t bN = (size_t)bI * Nc;
  size_t tbase = e0;                             // flat edge row base

  sT[wv][lane] = tup[tbase * 2 + lane];          // (i,j) pairs for 16 edges
  wave_sync_lds();

  _Float16* a1 = sA1[wv];
  _Float16* a2 = sA2[wv];

  // A1: [a_i | a_j | 0-pad] (16 x 64)
  for (int idx = lane; idx < 16 * 64; idx += 32) {
    int m = idx >> 6, c = idx & 63;
    float v = 0.f;
    if (c < 25)       v = au[(bN + sT[wv][m * 2 + 0]) * AT + c];
    else if (c < 50)  v = au[(bN + sT[wv][m * 2 + 1]) * AT + (c - 25)];
    st_a_sw(a1, m, c, (_Float16)v);
  }
  // A2 cols 50..99 = bond, cols >=100 = 0 (cols <50 filled after GEMM1)
  for (int idx = lane; idx < 16 * 128; idx += 32) {
    int m = idx >> 7, c = idx & 127;
    if (c < 50) continue;
    float v = (c < 100) ? bond[(tbase + m) * BD + (c - 50)] : 0.f;
    st_a_sw(a2, m, c, (_Float16)v);
  }
  wave_sync_lds();

  v8f acc1[4] = {};
#pragma unroll
  for (int ks = 0; ks < 2; ++ks) {
    v16h a = *(const v16h*)(a1 + (ks * 32 + lane) * 16);
#pragma unroll
    for (int nt = 0; nt < 4; ++nt) {
      v16h bb = *(const v16h*)(sWe + ((ks * 4 + nt) * 32 + lane) * 16);
      acc1[nt] = wmma_f16(a, bb, acc1[nt]);
    }
  }

  int nn = lane & 15, hb = lane >> 4;
#pragma unroll
  for (int nt = 0; nt < 4; ++nt) {
    int col = nt * 16 + nn;
    if (col < BD) {
      float bv = bemb[col];
#pragma unroll
      for (int v = 0; v < 8; ++v)
        st_a_sw(a2, v + hb * 8, col, (_Float16)softplusf(acc1[nt][v] + bv));
    }
  }
  wave_sync_lds();

  v8f acc2[4] = {};
#pragma unroll
  for (int ks = 0; ks < 4; ++ks) {
    v16h a = *(const v16h*)(a2 + (ks * 32 + lane) * 16);
#pragma unroll
    for (int nt = 0; nt < 4; ++nt) {
      v16h bb = *(const v16h*)(sWb + ((ks * 4 + nt) * 32 + lane) * 16);
      acc2[nt] = wmma_f16(a, bb, acc2[nt]);
    }
  }

#pragma unroll
  for (int nt = 0; nt < 4; ++nt) {
    int col = nt * 16 + nn;
    if (col < BD) {
      float bv = bbc[col];
#pragma unroll
      for (int v = 0; v < 8; ++v)
        out_bond[(tbase + v + hb * 8) * BD + col] = softplusf(acc2[nt][v] + bv);
    }
  }
}

// ---------------------------------------------------------------------------
extern "C" void kernel_launch(void* const* d_in, const int* in_sizes, int n_in,
                              void* d_out, int out_size, void* d_ws, size_t ws_size,
                              hipStream_t stream) {
  const float* atom = (const float*)d_in[0];
  const float* bond = (const float*)d_in[1];
  const int*   adj  = (const int*)d_in[2];
  const int*   tup  = (const int*)d_in[3];
  const float* Wta  = (const float*)d_in[4];
  const float* bta  = (const float*)d_in[5];
  const float* Wtb  = (const float*)d_in[6];
  const float* btb  = (const float*)d_in[7];
  const float* Wac  = (const float*)d_in[8];
  const float* bac  = (const float*)d_in[9];
  const float* Wemb = (const float*)d_in[10];
  const float* bemb = (const float*)d_in[11];
  const float* Wbc  = (const float*)d_in[12];
  const float* bbc  = (const float*)d_in[13];

  float* out_atom = (float*)d_out;                              // [B*N*25]
  float* out_bond = out_atom + (size_t)Bc * Nc * AT;            // [B*N*M*50]
  float* at_ws    = (float*)d_ws;                               // [B*N*25] f32

  // K1: 8192 row-tiles, 8 waves/block
  k_atomtype<<<dim3((Bc * Nc / 16) / 8), dim3(256), 0, stream>>>(
      atom, Wta, bta, Wtb, btb, at_ws);
  // K2: 32768 4-atom units, 8 waves/block; writes atom_update into d_out
  k_atomconv<<<dim3((Bc * Nc / 4) / 8), dim3(256), 0, stream>>>(
      at_ws, bond, adj, Wac, bac, out_atom);
  // K3: 98304 16-edge tiles, 4 waves/block
  k_bond<<<dim3((Bc * Nc * Mc / 16) / 4), dim3(128), 0, stream>>>(
      out_atom, bond, tup, Wemb, bemb, Wbc, bbc, out_bond);
}
